// RoPEMultiheadSelfAttention_59133109731618
// MI455X (gfx1250) — matrix operand order repaired and checked by reading, not compiled, not run on hardware
//
#include <hip/hip_runtime.h>
#include <hip/hip_bf16.h>
#include <stdint.h>

#define DIM    1024
#define NHEAD  16
#define HD     64
#define BATCH  4
#define SEQ    2048
#define BN     (BATCH * SEQ)    // 8192 rows
#define QKV_N  (3 * DIM)        // 3072 cols

typedef __bf16 bf16_t;
typedef __bf16 v16bf __attribute__((ext_vector_type(16)));
typedef __bf16 v8bf  __attribute__((ext_vector_type(8)));
typedef float  v8f   __attribute__((ext_vector_type(8)));

union FragA { v16bf v; v8bf h[2]; };

// ---------------- CDNA5 async global->LDS path (guarded) ----------------
#if defined(__has_builtin)
# if __has_builtin(__builtin_amdgcn_global_load_async_to_lds_b128)
#  define HAVE_ASYNC 1
# endif
#endif
#ifndef HAVE_ASYNC
# define HAVE_ASYNC 0
#endif

typedef int i32x4 __attribute__((vector_size(16)));
typedef __attribute__((address_space(1))) i32x4 as1_i32x4;
typedef __attribute__((address_space(3))) i32x4 as3_i32x4;

__device__ __forceinline__ void async_cp16(const bf16_t* g, bf16_t* l) {
#if HAVE_ASYNC
  __builtin_amdgcn_global_load_async_to_lds_b128(
      (as1_i32x4*)g, (as3_i32x4*)l, 0, 0);
#else
  *(v8bf*)l = *(const v8bf*)g;
#endif
}

__device__ __forceinline__ void wait_async_0() {
#if HAVE_ASYNC
# if __has_builtin(__builtin_amdgcn_s_wait_asynccnt)
  __builtin_amdgcn_s_wait_asynccnt(0);
# else
  asm volatile("s_wait_asynccnt 0" ::: "memory");
# endif
#endif
}

// ---------------- f32 -> bf16 convert ----------------
__global__ __launch_bounds__(256) void cvt_f32_bf16(const float* __restrict__ in,
                                                    bf16_t* __restrict__ out, int n) {
  int i = (blockIdx.x * 256 + threadIdx.x) * 4;
  if (i + 3 < n) {
    float4 f = *(const float4*)(in + i);
    out[i + 0] = (bf16_t)f.x;
    out[i + 1] = (bf16_t)f.y;
    out[i + 2] = (bf16_t)f.z;
    out[i + 3] = (bf16_t)f.w;
  }
}

// ---------------- WMMA bf16 NT GEMM: C[M,Nc] = A[M,K] * W[Nc,K]^T + bias --------
// Block tile 128x64, 4 waves, each wave 32x64 (2 A-frags x 4 B-frags = 8 WMMA/step).
// mode 0: scatter QKV epilogue (bf16 Q,K [B,H,N,hd], V transposed [B,H,hd,N])
// mode 1: plain f32 output [M, DIM]
__global__ __launch_bounds__(128)
void gemm_bf16_nt(const bf16_t* __restrict__ A, const bf16_t* __restrict__ W,
                  const float* __restrict__ bias, int K, int mode,
                  float* __restrict__ outF, bf16_t* __restrict__ Qr,
                  bf16_t* __restrict__ Kr, bf16_t* __restrict__ Vt) {
  __shared__ bf16_t Ab[2][128 * 32];   // 16 KB
  __shared__ bf16_t Bb[2][64 * 32];    //  8 KB

  const int t = threadIdx.x;
  const int wave = t >> 5;
  const int lane = t & 31;
  const int r16 = lane & 15;
  const int hiLo = (lane < 16) ? 0 : 8;
  const int mBase = blockIdx.x * 128;
  const int nBase = blockIdx.y * 64;

  // ---- precomputed staging addresses: row stride pattern, seg fixed per thread
  const int rB = t >> 2, seg = t & 3;  // 128 threads -> rows 0..31, segs 0..3
  const bf16_t* gA = A + (size_t)(mBase + rB) * K + seg * 8;
  const bf16_t* gW = W + (size_t)(nBase + rB) * K + seg * 8;
  bf16_t* lA = &Ab[0][rB * 32 + seg * 8];
  bf16_t* lB = &Bb[0][rB * 32 + seg * 8];

  auto stage = [&](int buf, int kOff) {
#pragma unroll
    for (int j = 0; j < 4; ++j)          // A: 128 rows, 32 rows per pass
      async_cp16(gA + (size_t)j * 32 * K + kOff,
                 lA + buf * (128 * 32) + j * 32 * 32);
#pragma unroll
    for (int j = 0; j < 2; ++j)          // W: 64 rows
      async_cp16(gW + (size_t)j * 32 * K + kOff,
                 lB + buf * (64 * 32) + j * 32 * 32);
  };

  v8f acc[2][4];
#pragma unroll
  for (int h = 0; h < 2; ++h)
    for (int c = 0; c < 4; ++c)
      for (int r = 0; r < 8; ++r) acc[h][c][r] = 0.f;

  // per-lane LDS read offsets (loop invariant)
  const int aOff0 = (wave * 32 + r16) * 32 + hiLo;
  const int aOff1 = (wave * 32 + 16 + r16) * 32 + hiLo;
  const int bOffL = r16 * 32 + hiLo;

  auto compute = [&](int buf) {
    const bf16_t* ab = &Ab[buf][0];
    const bf16_t* bb = &Bb[buf][0];
    FragA a0, a1, bf0, bf1, bf2, bf3;
    a0.h[0] = *(const v8bf*)(ab + aOff0);
    a0.h[1] = *(const v8bf*)(ab + aOff0 + 16);
    a1.h[0] = *(const v8bf*)(ab + aOff1);
    a1.h[1] = *(const v8bf*)(ab + aOff1 + 16);
    bf0.h[0] = *(const v8bf*)(bb + bOffL);
    bf0.h[1] = *(const v8bf*)(bb + bOffL + 16);
    bf1.h[0] = *(const v8bf*)(bb + bOffL + 16 * 32);
    bf1.h[1] = *(const v8bf*)(bb + bOffL + 16 * 32 + 16);
    bf2.h[0] = *(const v8bf*)(bb + bOffL + 32 * 32);
    bf2.h[1] = *(const v8bf*)(bb + bOffL + 32 * 32 + 16);
    bf3.h[0] = *(const v8bf*)(bb + bOffL + 48 * 32);
    bf3.h[1] = *(const v8bf*)(bb + bOffL + 48 * 32 + 16);
    acc[0][0] = __builtin_amdgcn_wmma_f32_16x16x32_bf16(false, a0.v, false, bf0.v, (short)0, acc[0][0], false, false);
    acc[0][1] = __builtin_amdgcn_wmma_f32_16x16x32_bf16(false, a0.v, false, bf1.v, (short)0, acc[0][1], false, false);
    acc[0][2] = __builtin_amdgcn_wmma_f32_16x16x32_bf16(false, a0.v, false, bf2.v, (short)0, acc[0][2], false, false);
    acc[0][3] = __builtin_amdgcn_wmma_f32_16x16x32_bf16(false, a0.v, false, bf3.v, (short)0, acc[0][3], false, false);
    acc[1][0] = __builtin_amdgcn_wmma_f32_16x16x32_bf16(false, a1.v, false, bf0.v, (short)0, acc[1][0], false, false);
    acc[1][1] = __builtin_amdgcn_wmma_f32_16x16x32_bf16(false, a1.v, false, bf1.v, (short)0, acc[1][1], false, false);
    acc[1][2] = __builtin_amdgcn_wmma_f32_16x16x32_bf16(false, a1.v, false, bf2.v, (short)0, acc[1][2], false, false);
    acc[1][3] = __builtin_amdgcn_wmma_f32_16x16x32_bf16(false, a1.v, false, bf3.v, (short)0, acc[1][3], false, false);
  };

  const int nsteps = K >> 5;   // K chunks of 32 (even: K=1024 -> 32)
  stage(0, 0);
  wait_async_0();
  __syncthreads();

  for (int s = 0; s < nsteps; s += 2) {
    if (s + 1 < nsteps) stage(1, (s + 1) * 32);
    compute(0);
    wait_async_0();
    __syncthreads();
    if (s + 2 < nsteps) stage(0, (s + 2) * 32);
    if (s + 1 < nsteps) compute(1);
    wait_async_0();
    __syncthreads();
  }

  // ---- epilogue (C layout: VGPR r -> row r|r+8; lane -> col) ----
  const int rAdd = (lane < 16) ? 0 : 8;
#pragma unroll
  for (int h2 = 0; h2 < 2; ++h2) {
#pragma unroll
    for (int c = 0; c < 4; ++c) {
      const int j = nBase + c * 16 + r16;
      const float bj = bias[j];
#pragma unroll
      for (int r = 0; r < 8; ++r) {
        const int m = mBase + wave * 32 + h2 * 16 + r + rAdd;
        const float val = acc[h2][c][r] + bj;
        if (mode == 1) {
          outF[(size_t)m * DIM + j] = val;
        } else {
          const int b = m >> 11, n = m & (SEQ - 1);
          const int which = j >> 10, d = j & (DIM - 1);
          const int h = d >> 6, e = d & (HD - 1);
          const bf16_t bv = (bf16_t)val;
          if (which == 0)
            Qr[(((size_t)b * NHEAD + h) * SEQ + n) * HD + e] = bv;
          else if (which == 1)
            Kr[(((size_t)b * NHEAD + h) * SEQ + n) * HD + e] = bv;
          else
            Vt[(((size_t)b * NHEAD + h) * HD + e) * SEQ + n] = bv;
        }
      }
    }
  }
}

// ---------------- RoPE over Q,K [B,H,N,hd] ----------------
__global__ __launch_bounds__(256)
void rope_kernel(bf16_t* __restrict__ Q, bf16_t* __restrict__ Kr) {
  const int t = threadIdx.x;
  const int i = t & 31;
  const size_t row = (size_t)blockIdx.x * 8 + (t >> 5);
  const int pos = (int)(row & (SEQ - 1));
  const float ang = (float)pos * __powf(10000.f, -(float)i * (1.f / 32.f));
  float s, c;
  __sincosf(ang, &s, &c);
  const size_t base = row * HD;
  float q0 = (float)Q[base + i], q1 = (float)Q[base + i + 32];
  Q[base + i]      = (bf16_t)(q0 * c - q1 * s);
  Q[base + i + 32] = (bf16_t)(q1 * c + q0 * s);
  float k0 = (float)Kr[base + i], k1 = (float)Kr[base + i + 32];
  Kr[base + i]      = (bf16_t)(k0 * c - k1 * s);
  Kr[base + i + 32] = (bf16_t)(k1 * c + k0 * s);
}

// ---------------- flash attention: 1 wave per 16-query tile ----------------
__global__ __launch_bounds__(32)
void attn_kernel(const bf16_t* __restrict__ Q, const bf16_t* __restrict__ Kmat,
                 const bf16_t* __restrict__ Vt, bf16_t* __restrict__ Out) {
  __shared__ bf16_t Pb[16 * 32];
  const int lane = threadIdx.x;
  const int r16 = lane & 15;
  const bool lo = lane < 16;
  const int qtile = blockIdx.x & 127;
  const int bh = blockIdx.x >> 7;
  const int qbase = qtile * 16;

  FragA qf[2];
  {
    const bf16_t* qrow = Q + ((size_t)bh * SEQ + qbase + r16) * HD;
    const int off = lo ? 0 : 8;
    qf[0].h[0] = *(const v8bf*)(qrow + off);
    qf[0].h[1] = *(const v8bf*)(qrow + off + 16);
    qf[1].h[0] = *(const v8bf*)(qrow + off + 32);
    qf[1].h[1] = *(const v8bf*)(qrow + off + 48);
  }

  float m[8], l[8];
  v8f acc[4];
  for (int r = 0; r < 8; ++r) { m[r] = -3.0e38f; l[r] = 0.f; }
  for (int c = 0; c < 4; ++c)
    for (int r = 0; r < 8; ++r) acc[c][r] = 0.f;

  const float scale = 0.125f;
  const bf16_t* Kb = Kmat + (size_t)bh * SEQ * HD;
  const bf16_t* Vb = Vt + (size_t)bh * HD * SEQ;

  for (int kb = 0; kb < SEQ; kb += 32) {
    v8f s01[2];
#pragma unroll
    for (int half = 0; half < 2; ++half) {
      FragA kf0, kf1;
      const bf16_t* krow = Kb + (size_t)(kb + half * 16 + r16) * HD + (lo ? 0 : 8);
      __builtin_prefetch(krow + 32 * HD, 0, 1);
      kf0.h[0] = *(const v8bf*)(krow);
      kf0.h[1] = *(const v8bf*)(krow + 16);
      kf1.h[0] = *(const v8bf*)(krow + 32);
      kf1.h[1] = *(const v8bf*)(krow + 48);
      v8f z;
      for (int r = 0; r < 8; ++r) z[r] = 0.f;
      z = __builtin_amdgcn_wmma_f32_16x16x32_bf16(false, qf[0].v, false, kf0.v,
                                                  (short)0, z, false, false);
      z = __builtin_amdgcn_wmma_f32_16x16x32_bf16(false, qf[1].v, false, kf1.v,
                                                  (short)0, z, false, false);
      s01[half] = z;
    }

    float p0[8], p1[8];
#pragma unroll
    for (int r = 0; r < 8; ++r) {
      float a0 = s01[0][r] * scale, a1 = s01[1][r] * scale;
      float mx = fmaxf(a0, a1);
      for (int msk = 1; msk < 16; msk <<= 1)
        mx = fmaxf(mx, __shfl_xor(mx, msk, 32));
      const float mn = fmaxf(m[r], mx);
      const float e0 = __expf(a0 - mn), e1 = __expf(a1 - mn);
      float sum = e0 + e1;
      for (int msk = 1; msk < 16; msk <<= 1) sum += __shfl_xor(sum, msk, 32);
      const float corr = __expf(m[r] - mn);
      l[r] = l[r] * corr + sum;
      m[r] = mn;
      for (int c = 0; c < 4; ++c) acc[c][r] *= corr;
      p0[r] = e0; p1[r] = e1;
    }

    __syncthreads();
    {
      const int rowOff = lo ? 0 : 8;
#pragma unroll
      for (int r = 0; r < 8; ++r) {
        Pb[(r + rowOff) * 32 + r16]      = (bf16_t)p0[r];
        Pb[(r + rowOff) * 32 + 16 + r16] = (bf16_t)p1[r];
      }
    }
    __syncthreads();
    FragA pf;
    {
      const int off = lo ? 0 : 8;
      pf.h[0] = *(const v8bf*)&Pb[r16 * 32 + off];
      pf.h[1] = *(const v8bf*)&Pb[r16 * 32 + off + 16];
    }

#pragma unroll
    for (int c = 0; c < 4; ++c) {
      FragA vf;
      const bf16_t* vrow = Vb + (size_t)(c * 16 + r16) * SEQ + kb + (lo ? 0 : 8);
      vf.h[0] = *(const v8bf*)(vrow);
      vf.h[1] = *(const v8bf*)(vrow + 16);
      acc[c] = __builtin_amdgcn_wmma_f32_16x16x32_bf16(
          false, pf.v, false, vf.v, (short)0, acc[c], false, false);
    }
  }

  const int b = bh >> 4, h = bh & 15;
  const int rAdd = lo ? 0 : 8;
#pragma unroll
  for (int c = 0; c < 4; ++c)
#pragma unroll
    for (int r = 0; r < 8; ++r) {
      const int n = qbase + r + rAdd;
      const int dcol = h * HD + c * 16 + r16;
      Out[((size_t)b * SEQ + n) * DIM + dcol] = (bf16_t)(acc[c][r] / l[r]);
    }
}

// ---------------- launch ----------------
extern "C" void kernel_launch(void* const* d_in, const int* in_sizes, int n_in,
                              void* d_out, int out_size, void* d_ws, size_t ws_size,
                              hipStream_t stream) {
  const float* x      = (const float*)d_in[0];
  const float* qkv_w  = (const float*)d_in[1];
  const float* qkv_b  = (const float*)d_in[2];
  const float* proj_w = (const float*)d_in[3];
  const float* proj_b = (const float*)d_in[4];
  float* out = (float*)d_out;

  char* ws = (char*)d_ws;
  bf16_t* Xb  = (bf16_t*)ws; ws += (size_t)BN * DIM * 2;
  bf16_t* Wb  = (bf16_t*)ws; ws += (size_t)QKV_N * DIM * 2;
  bf16_t* Wpb = (bf16_t*)ws; ws += (size_t)DIM * DIM * 2;
  bf16_t* Qr  = (bf16_t*)ws; ws += (size_t)BATCH * NHEAD * SEQ * HD * 2;
  bf16_t* Kr  = (bf16_t*)ws; ws += (size_t)BATCH * NHEAD * SEQ * HD * 2;
  bf16_t* Vt  = (bf16_t*)ws; ws += (size_t)BATCH * NHEAD * SEQ * HD * 2;
  bf16_t* Ao  = (bf16_t*)ws;

  cvt_f32_bf16<<<(BN * DIM) / 1024, 256, 0, stream>>>(x, Xb, BN * DIM);
  cvt_f32_bf16<<<(QKV_N * DIM) / 1024, 256, 0, stream>>>(qkv_w, Wb, QKV_N * DIM);
  cvt_f32_bf16<<<(DIM * DIM) / 1024, 256, 0, stream>>>(proj_w, Wpb, DIM * DIM);

  gemm_bf16_nt<<<dim3(BN / 128, QKV_N / 64), 128, 0, stream>>>(
      Xb, Wb, qkv_b, DIM, 0, nullptr, Qr, Kr, Vt);

  rope_kernel<<<(BATCH * NHEAD * SEQ) / 8, 256, 0, stream>>>(Qr, Kr);

  attn_kernel<<<BATCH * NHEAD * (SEQ / 16), 32, 0, stream>>>(Qr, Kr, Vt, Ao);

  gemm_bf16_nt<<<dim3(BN / 128, DIM / 64), 128, 0, stream>>>(
      Ao, Wpb, proj_b, DIM, 1, out, nullptr, nullptr, nullptr);
}
